// MoELayer_12738873000187
// MI455X (gfx1250) — matrix operand order repaired and checked by reading, not compiled, not run on hardware
//
#include <hip/hip_runtime.h>
#include <hip/hip_bf16.h>

// ---------------- problem constants (match reference) ----------------
constexpr int B_TOK = 8192;
constexpr int E_EXP = 16;
constexpr int H_IN  = 1024;
constexpr int H_OUT = 1024;
constexpr int TOPK  = 4;

typedef __attribute__((ext_vector_type(16))) _Float16 v16h;
typedef __attribute__((ext_vector_type(8)))  _Float16 v8h;
typedef __attribute__((ext_vector_type(8)))  float    v8f;

// workspace layout (in 4-byte elements)
constexpr int WS_TOP4ID   = 64;
constexpr int WS_TOP4P    = WS_TOP4ID + B_TOK * 4;
constexpr int WS_TOKLIST  = WS_TOP4P  + B_TOK * 4;
constexpr int WS_PROBLIST = WS_TOKLIST + E_EXP * B_TOK;
constexpr int WS_END      = WS_PROBLIST + E_EXP * B_TOK;   // elements (16B-aligned byte offset)
constexpr size_t WE16_BYTES = (size_t)E_EXP * H_OUT * H_IN * 2;  // f16 W_e copy

// GEMM tiling
constexpr int M_T   = 64;          // tokens per block tile
constexpr int N_T   = 128;         // output cols per block tile (8 waves x 16)
constexpr int LDA_H = H_IN + 8;    // padded f16 row stride -> conflict-free ds_load_b128
constexpr size_t SMEM_BYTES = 512 + (size_t)M_T * LDA_H * sizeof(_Float16);

// ---------------------------------------------------------------------
// 1) Router: logits = x*W_r^T + b_r, top-4, softmax, build expert lists
// ---------------------------------------------------------------------
__global__ __launch_bounds__(128)
void router_kernel(const float* __restrict__ x,
                   const float* __restrict__ W_r,
                   const float* __restrict__ b_r,
                   int*   __restrict__ counts,
                   int*   __restrict__ selcnt,
                   float* __restrict__ fpacc,
                   int*   __restrict__ top4id,
                   float* __restrict__ top4p,
                   int*   __restrict__ toklist,
                   float* __restrict__ problist) {
    __shared__ float Wr[E_EXP * H_IN];  // 64 KB
    const int tid = threadIdx.x;

    const float4* wsrc = reinterpret_cast<const float4*>(W_r);
    float4*       wdst = reinterpret_cast<float4*>(Wr);
    for (int i = tid; i < E_EXP * H_IN / 4; i += blockDim.x) wdst[i] = wsrc[i];
    __syncthreads();

    const int b = blockIdx.x * blockDim.x + tid;

    float acc[E_EXP];
#pragma unroll
    for (int e = 0; e < E_EXP; ++e) acc[e] = b_r[e];

    const float4* xr = reinterpret_cast<const float4*>(x + (size_t)b * H_IN);
    for (int k4 = 0; k4 < H_IN / 4; ++k4) {
        float4 xv = xr[k4];
#pragma unroll
        for (int e = 0; e < E_EXP; ++e) {
            float4 wv = reinterpret_cast<const float4*>(Wr + e * H_IN)[k4];
            acc[e] += xv.x * wv.x + xv.y * wv.y + xv.z * wv.z + xv.w * wv.w;
        }
    }

    // top-4 (first-index wins on ties, like argmax/top_k)
    float l[E_EXP];
#pragma unroll
    for (int e = 0; e < E_EXP; ++e) l[e] = acc[e];
    int ids[TOPK]; float vals[TOPK];
#pragma unroll
    for (int k = 0; k < TOPK; ++k) {
        int   bi = 0;
        float bv = l[0];
#pragma unroll
        for (int e = 1; e < E_EXP; ++e)
            if (l[e] > bv) { bv = l[e]; bi = e; }
        ids[k] = bi; vals[k] = bv; l[bi] = -3.0e38f;
    }

    // softmax over survivors (masked experts are exactly zero)
    const float m = vals[0];
    float p[TOPK], s = 0.0f;
#pragma unroll
    for (int k = 0; k < TOPK; ++k) { p[k] = __expf(vals[k] - m); s += p[k]; }
    const float inv = 1.0f / s;
#pragma unroll
    for (int k = 0; k < TOPK; ++k) p[k] *= inv;

    atomicAdd(&selcnt[ids[0]], 1);
#pragma unroll
    for (int k = 0; k < TOPK; ++k) {
        top4id[b * 4 + k] = ids[k];
        top4p [b * 4 + k] = p[k];
        atomicAdd(&fpacc[ids[k]], p[k]);
        int pos = atomicAdd(&counts[ids[k]], 1);
        toklist [ids[k] * B_TOK + pos] = b;
        problist[ids[k] * B_TOK + pos] = p[k];
    }
}

// ---------------------------------------------------------------------
// 2) Initialize out with the prob-weighted biases
// ---------------------------------------------------------------------
__global__ __launch_bounds__(128)
void bias_out_kernel(const float* __restrict__ b_e,
                     const int*   __restrict__ top4id,
                     const float* __restrict__ top4p,
                     float* __restrict__ out) {
    const int b = blockIdx.x;
    int   ids[TOPK]; float ps[TOPK];
#pragma unroll
    for (int k = 0; k < TOPK; ++k) { ids[k] = top4id[b * 4 + k]; ps[k] = top4p[b * 4 + k]; }
    for (int c = threadIdx.x; c < H_OUT; c += blockDim.x) {
        float v = 0.0f;
#pragma unroll
        for (int k = 0; k < TOPK; ++k) v += ps[k] * b_e[ids[k] * H_OUT + c];
        out[(size_t)b * H_OUT + c] = v;
    }
}

// ---------------------------------------------------------------------
// 2b) One-time W_e f32 -> f16 pre-conversion (amortized across all tiles)
// ---------------------------------------------------------------------
__global__ __launch_bounds__(256)
void convert_we_kernel(const float* __restrict__ W_e, _Float16* __restrict__ W16) {
    const size_t i = ((size_t)blockIdx.x * blockDim.x + threadIdx.x) * 8;
    float4 a = *reinterpret_cast<const float4*>(W_e + i);
    float4 b = *reinterpret_cast<const float4*>(W_e + i + 4);
    v8h h;
    h[0] = (_Float16)a.x; h[1] = (_Float16)a.y; h[2] = (_Float16)a.z; h[3] = (_Float16)a.w;
    h[4] = (_Float16)b.x; h[5] = (_Float16)b.y; h[6] = (_Float16)b.z; h[7] = (_Float16)b.w;
    *reinterpret_cast<v8h*>(W16 + i) = h;
}

// ---------------------------------------------------------------------
// 3) Grouped gather-GEMM: 64-token x 128-col tiles, 4 WMMA per B-frag
// ---------------------------------------------------------------------
template <bool PRECONV>
__global__ __launch_bounds__(256)
void expert_gemm_kernel(const float*    __restrict__ x,
                        const float*    __restrict__ W_e,
                        const _Float16* __restrict__ W16,
                        const int*      __restrict__ counts,
                        const int*      __restrict__ toklist,
                        const float*    __restrict__ problist,
                        float* __restrict__ out) {
    const int e    = blockIdx.z;
    const int cnt  = counts[e];
    const int row0 = blockIdx.x * M_T;
    if (row0 >= cnt) return;
    const int rows = min(M_T, cnt - row0);

    extern __shared__ char smem[];
    int*      tokS  = reinterpret_cast<int*>(smem);                 // 64 ints
    float*    probS = reinterpret_cast<float*>(smem + 256);         // 64 floats
    _Float16* As    = reinterpret_cast<_Float16*>(smem + 512);      // 64 x LDA_H

    const int tid = threadIdx.x;
    if (tid < M_T) {
        if (tid < rows) {
            tokS [tid] = toklist [e * B_TOK + row0 + tid];
            probS[tid] = problist[e * B_TOK + row0 + tid];
        } else { tokS[tid] = 0; probS[tid] = 0.0f; }
    }
    __syncthreads();

    // gather + convert A tile: 256 threads x 4 cols each, all 64 rows
    const int c0 = tid * 4;
    for (int r = 0; r < M_T; ++r) {
        float4 v = make_float4(0.f, 0.f, 0.f, 0.f);
        if (r < rows)
            v = *reinterpret_cast<const float4*>(x + (size_t)tokS[r] * H_IN + c0);
        _Float16* dst = &As[r * LDA_H + c0];
        dst[0] = (_Float16)v.x; dst[1] = (_Float16)v.y;
        dst[2] = (_Float16)v.z; dst[3] = (_Float16)v.w;
    }
    __syncthreads();

    const int wave = tid >> 5;
    const int lane = tid & 31;
    const int hsel = lane >> 4;   // half-wave select
    const int lrow = lane & 15;   // A row / B column within 16x16 tile
    const int bcol = blockIdx.y * N_T + wave * 16 + lrow;
    const size_t wofs = ((size_t)e * H_OUT + bcol) * H_IN;
    const int aoff = hsel * 16;   // A: lanes 0-15 hold K 0..15, lanes 16-31 K 16..31
    const int boff = hsel * 16;   // B: lanes 0-15 hold K 0..15, lanes 16-31 K 16..31

    v8f acc[4] = {{}, {}, {}, {}};
    for (int kb = 0; kb < H_IN; kb += 32) {
        // ---- B fragment: one per wave per K-step, reused by 4 WMMAs ----
        v16h bf;
        if constexpr (PRECONV) {
            bf = *reinterpret_cast<const v16h*>(W16 + wofs + kb + boff);
        } else {
            const float* bp = W_e + wofs + kb + boff;
#pragma unroll
            for (int i = 0; i < 16; i += 4) {
                float4 f = *reinterpret_cast<const float4*>(bp + i);
                bf[i]     = (_Float16)f.x; bf[i + 1] = (_Float16)f.y;
                bf[i + 2] = (_Float16)f.z; bf[i + 3] = (_Float16)f.w;
            }
        }
        // ---- load all 4 A fragments first (one clause, one wait) ----
        v16h afr[4];
#pragma unroll
        for (int mt = 0; mt < 4; ++mt) {
            const _Float16* ab = &As[(mt * 16 + lrow) * LDA_H + kb + aoff];
            v8h alo = *reinterpret_cast<const v8h*>(ab);
            v8h ahi = *reinterpret_cast<const v8h*>(ab + 8);
#pragma unroll
            for (int i = 0; i < 8; ++i) { afr[mt][i] = alo[i]; afr[mt][i + 8] = ahi[i]; }
        }
        // ---- 4 back-to-back WMMAs (independent D, shared B: no hazards) ----
#pragma unroll
        for (int mt = 0; mt < 4; ++mt)
            acc[mt] = __builtin_amdgcn_wmma_f32_16x16x32_f16(
                false, afr[mt], false, bf, (short)0, acc[mt], false, false);
    }

    // prob-scaled scatter-accumulate: C VGPR r holds M = r + 8*hsel, N = lane%16
    float* outc = out + bcol;
    if (rows == M_T) {
        // full tile (the overwhelmingly common case): no per-element exec masking
#pragma unroll
        for (int mt = 0; mt < 4; ++mt) {
#pragma unroll
            for (int r = 0; r < 8; ++r) {
                const int m = mt * 16 + r + hsel * 8;
                atomicAdd(outc + (size_t)tokS[m] * H_OUT, probS[m] * acc[mt][r]);
            }
        }
    } else {
#pragma unroll
        for (int mt = 0; mt < 4; ++mt) {
#pragma unroll
            for (int r = 0; r < 8; ++r) {
                const int m = mt * 16 + r + hsel * 8;
                if (m < rows)
                    atomicAdd(outc + (size_t)tokS[m] * H_OUT, probS[m] * acc[mt][r]);
            }
        }
    }
}

// ---------------------------------------------------------------------
// 4) frac_prob + load-balance loss into the output tail
// ---------------------------------------------------------------------
__global__ void finalize_kernel(const int*   __restrict__ selcnt,
                                const float* __restrict__ fpacc,
                                float* __restrict__ out_tail) {
    __shared__ float fp[E_EXP];
    const int e = threadIdx.x;
    if (e < E_EXP) {
        fp[e] = fpacc[e] * (1.0f / (float)B_TOK);
        out_tail[e] = fp[e];
    }
    __syncthreads();
    if (e == 0) {
        float loss = 0.0f;
        for (int j = 0; j < E_EXP; ++j)
            loss += ((float)selcnt[j] * (1.0f / (float)B_TOK)) * fp[j];
        out_tail[E_EXP] = (float)E_EXP * loss;
    }
}

// ---------------------------------------------------------------------
extern "C" void kernel_launch(void* const* d_in, const int* in_sizes, int n_in,
                              void* d_out, int out_size, void* d_ws, size_t ws_size,
                              hipStream_t stream) {
    (void)in_sizes; (void)n_in; (void)out_size;
    const float* x   = (const float*)d_in[0];
    const float* W_e = (const float*)d_in[1];
    const float* b_e = (const float*)d_in[2];
    const float* W_r = (const float*)d_in[3];
    const float* b_r = (const float*)d_in[4];
    float* out = (float*)d_out;

    int*   wsI = (int*)d_ws;
    float* wsF = (float*)d_ws;
    int*   counts   = wsI;
    int*   selcnt   = wsI + 16;
    float* fpacc    = wsF + 32;
    int*   top4id   = wsI + WS_TOP4ID;
    float* top4p    = wsF + WS_TOP4P;
    int*   toklist  = wsI + WS_TOKLIST;
    float* problist = wsF + WS_PROBLIST;
    _Float16* W16   = reinterpret_cast<_Float16*>((char*)d_ws + (size_t)WS_END * 4);

    const bool preconv = ws_size >= (size_t)WS_END * 4 + WE16_BYTES;

    hipMemsetAsync(d_ws, 0, 64 * sizeof(int), stream);

    router_kernel<<<B_TOK / 128, 128, 0, stream>>>(
        x, W_r, b_r, counts, selcnt, fpacc, top4id, top4p, toklist, problist);

    bias_out_kernel<<<B_TOK, 128, 0, stream>>>(b_e, top4id, top4p, out);

    const dim3 ggrid(B_TOK / M_T, H_OUT / N_T, E_EXP);
    if (preconv) {
        convert_we_kernel<<<(int)((size_t)E_EXP * H_OUT * H_IN / 8 / 256), 256, 0, stream>>>(W_e, W16);
        expert_gemm_kernel<true><<<ggrid, 256, SMEM_BYTES, stream>>>(
            x, W_e, W16, counts, toklist, problist, out);
    } else {
        expert_gemm_kernel<false><<<ggrid, 256, SMEM_BYTES, stream>>>(
            x, W_e, W16, counts, toklist, problist, out);
    }

    finalize_kernel<<<1, 32, 0, stream>>>(selcnt, fpacc, out + (size_t)B_TOK * H_OUT);
}
